// Graformer_27797028339854
// MI455X (gfx1250) — matrix operand order repaired and checked by reading, not compiled, not run on hardware
//
#include <hip/hip_runtime.h>
#include <hip/hip_bf16.h>

// Graformer layer for MI455X (gfx1250, wave32, WMMA).
// - GEMMs: v_wmma_f32_16x16x32_f16, f16 operands / f32 accum, 64x64 block tile,
//   double-buffered LDS, async global->LDS staging via
//   GLOBAL_LOAD_ASYNC_TO_LDS_B128 (ASYNCcnt), VGPR-staged fallback otherwise.
// - Edge segment-softmax: ordered-uint atomic max + float atomic adds.
// Roofline: ~1.4e11 dense FLOPs but ~1-1.5 GB of HBM traffic dominates
// (23.3 TB/s -> ~50-65 us floor), so f16 operand staging + latency hiding
// are the levers; WMMA keeps the GEMMs far off the critical path.

typedef __attribute__((ext_vector_type(16))) _Float16 v16h;
typedef __attribute__((ext_vector_type(8)))  float    v8f;
typedef __attribute__((ext_vector_type(4)))  int      v4i;

#define LDA_S 48   // 32 + 16 pad (f16) -> 96 B rows, 16B-aligned for b128
#define LDB_S 80   // 64 + 16 pad (f16) -> 160 B rows, 16B-aligned for b128

#if defined(__AMDGCN__) && __has_builtin(__builtin_amdgcn_global_load_async_to_lds_b128)
#define USE_ASYNC_LDS 1
typedef __attribute__((address_space(1))) v4i gas_v4i;   // global (prints as __device__)
typedef __attribute__((address_space(3))) v4i las_v4i;   // LDS    (prints as __shared__)
#endif

__device__ __forceinline__ void wait_async_lds() {
#if defined(USE_ASYNC_LDS)
#if __has_builtin(__builtin_amdgcn_s_wait_asynccnt)
    __builtin_amdgcn_s_wait_asynccnt(0);
#else
    asm volatile("s_wait_asynccnt 0" ::: "memory");
#endif
#endif
}

// ---------------------------------------------------------------- fragments
__device__ __forceinline__ v16h load_frag_a(const _Float16* p) {
    // A: 16x32 f16, ISA layout: lanes 0-15 rows M, K in {0..7,16..23};
    // lanes 16-31 same rows, K in {8..15,24..31}; VGPR j holds a K-pair.
    const int lane = threadIdx.x & 31;
    const int half = lane >> 4;
    const int m    = lane & 15;
    const _Float16* r = p + m * LDA_S;
    v16h a;
#pragma unroll
    for (int j = 0; j < 8; ++j) {
        const int kb = ((j < 4) ? (2 * j) : (16 + 2 * (j - 4))) + half * 8;
        union { unsigned u; _Float16 h[2]; } cv;
        cv.u = *(const unsigned*)(r + kb);
        a[2 * j]     = cv.h[0];
        a[2 * j + 1] = cv.h[1];
    }
    return a;
}

__device__ __forceinline__ v16h load_frag_b(const _Float16* p) {
    // B: 32x16 f16 (KxN): lane l holds col N=l%16; K in {0..7,16..23} (lanes 0-15), {8..15,24..31} (lanes 16-31).
    const int lane = threadIdx.x & 31;
    const int half = lane >> 4;
    const _Float16* r = p + (lane & 15);
    v16h b;
#pragma unroll
    for (int s = 0; s < 16; ++s) {
        const int kk = ((s < 8) ? s : (s + 8)) + half * 8;
        b[s] = r[kk * LDB_S];
    }
    return b;
}

// ---------------------------------------------------------------- WMMA GEMM
// C[MxN] = A[MxK] * B[KxN], A/B f16 row-major, accum f32.
// MODE 0: Cf = C                     (plain fp32 store)
// MODE 1: Cf = prev + scale[col]*C   (fused residual, fp32 store)
// MODE 2: Ch = gelu(C)               (exact gelu, f16 store)
template <int MODE>
__launch_bounds__(256)
__global__ void gemm_f16_wmma(const _Float16* __restrict__ A,
                              const _Float16* __restrict__ B,
                              int M, int N, int K,
                              float* __restrict__ Cf,
                              _Float16* __restrict__ Ch,
                              const float* __restrict__ prev,
                              const float* __restrict__ scale) {
    __shared__ __align__(16) _Float16 As[2][64 * LDA_S];
    __shared__ __align__(16) _Float16 Bs[2][32 * LDB_S];

    const int t    = threadIdx.x;
    const int lane = t & 31;
    const int wid  = t >> 5;
    const int wm   = wid & 3;   // 4 row strips of 16
    const int wn   = wid >> 2;  // 2 col halves of 32
    const int m0   = blockIdx.x * 64;
    const int n0   = blockIdx.y * 64;

    v8f acc0 = {}; v8f acc1 = {};

    // staging assignment: one 16B chunk per thread per tile
    const int arow = t >> 2, acg = (t & 3) * 8;   // A: 64 rows x 32 cols
    const int brow = t >> 3, bcg = (t & 7) * 8;   // B: 32 rows x 64 cols
    int garow = m0 + arow; if (garow > M - 1) garow = M - 1;  // clamp; stores guarded

    const _Float16* gA = A + (size_t)garow * K + acg;
    const _Float16* gB = B + (size_t)brow * N + n0 + bcg;

    auto stage = [&](int k0, int buf) {
#if defined(USE_ASYNC_LDS)
        __builtin_amdgcn_global_load_async_to_lds_b128(
            (gas_v4i*)(gA + k0),
            (las_v4i*)&As[buf][arow * LDA_S + acg], 0, 0);
        __builtin_amdgcn_global_load_async_to_lds_b128(
            (gas_v4i*)(gB + (size_t)k0 * N),
            (las_v4i*)&Bs[buf][brow * LDB_S + bcg], 0, 0);
#else
        uint4 av = *(const uint4*)(gA + k0);
        uint4 bv = *(const uint4*)(gB + (size_t)k0 * N);
        *(uint4*)(&As[buf][arow * LDA_S + acg]) = av;
        *(uint4*)(&Bs[buf][brow * LDB_S + bcg]) = bv;
#endif
    };

    // prologue: fill buffer 0
    stage(0, 0);
    wait_async_lds();
    __syncthreads();

    int cur = 0;
    for (int k0 = 0; k0 < K; k0 += 32) {
        if (k0 + 32 < K) {
            stage(k0 + 32, cur ^ 1);           // overlap with WMMA below
#if !defined(USE_ASYNC_LDS)
            __builtin_prefetch(gA + k0 + 64, 0, 1);
            __builtin_prefetch(gB + (size_t)(k0 + 64) * N, 0, 1);
#endif
        }
        v16h af  = load_frag_a(&As[cur][wm * 16 * LDA_S]);
        v16h bf0 = load_frag_b(&Bs[cur][wn * 32]);
        v16h bf1 = load_frag_b(&Bs[cur][wn * 32 + 16]);
        acc0 = __builtin_amdgcn_wmma_f32_16x16x32_f16(false, af, false, bf0,
                                                      (short)0, acc0, false, false);
        acc1 = __builtin_amdgcn_wmma_f32_16x16x32_f16(false, af, false, bf1,
                                                      (short)0, acc1, false, false);
        wait_async_lds();    // my async writes into buf^1 have landed
        __syncthreads();     // everyone's have landed; readers of cur done
        cur ^= 1;
    }

    // epilogue: C/D layout — VGPR j: row j (lanes 0-15) / j+8 (lanes 16-31)
    const int half  = lane >> 4;
    const int nn    = lane & 15;
    const int rbase = m0 + wm * 16 + half * 8;
    const int cb    = n0 + wn * 32 + nn;
#pragma unroll
    for (int j = 0; j < 8; ++j) {
        const int row = rbase + j;
        if (row >= M) continue;
        const float c0 = acc0[j];
        const float c1 = acc1[j];
        const size_t o0 = (size_t)row * N + cb;
        const size_t o1 = o0 + 16;
        if (MODE == 0) {
            Cf[o0] = c0; Cf[o1] = c1;
        } else if (MODE == 1) {
            Cf[o0] = prev[o0] + scale[cb]      * c0;
            Cf[o1] = prev[o1] + scale[cb + 16] * c1;
        } else {
            Ch[o0] = (_Float16)(0.5f * c0 * (1.0f + erff(c0 * 0.70710678f)));
            Ch[o1] = (_Float16)(0.5f * c1 * (1.0f + erff(c1 * 0.70710678f)));
        }
    }
}

// ---------------------------------------------------------------- LayerNorm
// one wave per 256-elem row, f16 output (feeds WMMA GEMMs)
__launch_bounds__(256)
__global__ void ln_f16_kernel(const float* __restrict__ in,
                              const float* __restrict__ g,
                              const float* __restrict__ b,
                              _Float16* __restrict__ out, int V) {
    const int wid  = threadIdx.x >> 5;
    const int lane = threadIdx.x & 31;
    const int row  = blockIdx.x * 8 + wid;
    if (row >= V) return;
    const float* x = in + (size_t)row * 256;
    float v[8]; float s = 0.f;
#pragma unroll
    for (int i = 0; i < 8; ++i) { v[i] = x[lane + i * 32]; s += v[i]; }
#pragma unroll
    for (int off = 16; off; off >>= 1) s += __shfl_xor(s, off, 32);
    const float mean = s * (1.0f / 256.0f);
    float q = 0.f;
#pragma unroll
    for (int i = 0; i < 8; ++i) { const float d = v[i] - mean; q += d * d; }
#pragma unroll
    for (int off = 16; off; off >>= 1) q += __shfl_xor(q, off, 32);
    const float inv = rsqrtf(q * (1.0f / 256.0f) + 0.001f);
    _Float16* o = out + (size_t)row * 256;
#pragma unroll
    for (int i = 0; i < 8; ++i) {
        const int col = lane + i * 32;
        o[col] = (_Float16)((v[i] - mean) * inv * g[col] + b[col]);
    }
}

// ---------------------------------------------------------------- utilities
__global__ void cast_f32_to_f16(const float* __restrict__ in,
                                _Float16* __restrict__ out, long long n) {
    const long long i = (long long)blockIdx.x * 256 + threadIdx.x;
    if (i < n) out[i] = (_Float16)in[i];
}

__global__ void init_softmax_bufs(unsigned* __restrict__ smax,
                                  float* __restrict__ den,
                                  float* __restrict__ agg,
                                  long long nVH, long long nAgg) {
    const long long i = (long long)blockIdx.x * 256 + threadIdx.x;
    if (i < nVH) { smax[i] = 0u; den[i] = 0.f; }
    if (i < nAgg) agg[i] = 0.f;
}

// ---------------------------------------------------------------- edge pass 1
// scores[(l*E+e)*8+h] = <key,query> + add_bias ; atomic ordered-max per (tgt,h)
__launch_bounds__(256)
__global__ void edge_scores_kernel(const int* __restrict__ adj0,
                                   const int* __restrict__ adj1,
                                   const float* __restrict__ ef0,
                                   const float* __restrict__ ef1,
                                   const float* __restrict__ kqv,
                                   const float* __restrict__ multb,
                                   const float* __restrict__ addb,
                                   const float* __restrict__ efW,
                                   float* __restrict__ scores,
                                   unsigned* __restrict__ smax, int E) {
    __shared__ float efs[2][16];
    const int t   = threadIdx.x;
    const int sub = t >> 7;                 // 2 edges per block
    const int le  = blockIdx.x * 2 + sub;
    const int l   = le / E;
    const int e   = le - l * E;
    const int* adj = l ? adj1 : adj0;
    const int src = adj[2 * e];
    const int tgt = adj[2 * e + 1];
    const int idx = t & 127;                // h*16 + k
    const int hh  = idx >> 4;
    const int k   = idx & 15;

    const float* ef = (l ? ef1 : ef0) + (size_t)e * 16;
    if (idx < 16) efs[sub][idx] = ef[idx];
    __syncthreads();

    float key = kqv[(size_t)src * 768 + l * 384 + hh * 48 + k] + multb[l * 128 + idx];
    const float* w = efW + l * 2048 + idx;
#pragma unroll
    for (int d = 0; d < 16; ++d) key += efs[sub][d] * w[d * 128];

    const float q = kqv[(size_t)tgt * 768 + l * 384 + hh * 48 + 16 + k];
    float p = key * q;
#pragma unroll
    for (int off = 8; off; off >>= 1) p += __shfl_xor(p, off, 16);

    if (k == 0) {
        const float sc = p + addb[l * 8 + hh];
        scores[(size_t)le * 8 + hh] = sc;
        unsigned u = __float_as_uint(sc);
        u = (u & 0x80000000u) ? ~u : (u | 0x80000000u);   // order-preserving
        atomicMax(&smax[(size_t)tgt * 8 + hh], u);
    }
}

// ---------------------------------------------------------------- edge pass 2
__global__ void edge_exp_kernel(const int* __restrict__ adj0,
                                const int* __restrict__ adj1,
                                float* __restrict__ scores,
                                const unsigned* __restrict__ smax,
                                float* __restrict__ den, int E) {
    const long long i = (long long)blockIdx.x * 256 + threadIdx.x;
    if (i >= (long long)2 * E * 8) return;
    const int hh = (int)(i & 7);
    const int le = (int)(i >> 3);
    const int l  = le / E;
    const int e  = le - l * E;
    const int tgt = (l ? adj1 : adj0)[2 * e + 1];
    unsigned u = smax[(size_t)tgt * 8 + hh];
    u = (u & 0x80000000u) ? (u & 0x7FFFFFFFu) : ~u;       // decode
    const float m  = __uint_as_float(u);
    const float ex = __expf(scores[i] - m);
    scores[i] = ex;
    atomicAdd(&den[(size_t)tgt * 8 + hh], ex);
}

// ---------------------------------------------------------------- edge pass 3
__global__ void edge_agg_kernel(const int* __restrict__ adj0,
                                const int* __restrict__ adj1,
                                const float* __restrict__ ex,
                                const float* __restrict__ den,
                                const float* __restrict__ kqv,
                                float* __restrict__ agg, int E) {
    const long long i = (long long)blockIdx.x * 256 + threadIdx.x;
    if (i >= (long long)2 * E * 128) return;
    const int idx = (int)(i & 127);             // h*16 + m
    const int le  = (int)(i >> 7);
    const int l   = le / E;
    const int e   = le - l * E;
    const int* adj = l ? adj1 : adj0;
    const int src = adj[2 * e];
    const int tgt = adj[2 * e + 1];
    const int hh  = idx >> 4;
    const int mm  = idx & 15;
    const float attn = ex[(size_t)le * 8 + hh] / den[(size_t)tgt * 8 + hh];
    const float v = kqv[(size_t)src * 768 + l * 384 + hh * 48 + 32 + mm];
    atomicAdd(&agg[(size_t)tgt * 128 + idx], attn * v);
}

// ---------------------------------------------------------------- launcher
static inline size_t ws_align(size_t x) { return (x + 255) & ~(size_t)255; }

extern "C" void kernel_launch(void* const* d_in, const int* in_sizes, int n_in,
                              void* d_out, int out_size, void* d_ws, size_t ws_size,
                              hipStream_t stream) {
    const float* h_emb = (const float*)d_in[0];
    const int*   adj0  = (const int*)d_in[1];
    const int*   adj1  = (const int*)d_in[2];
    const float* ef0   = (const float*)d_in[3];
    const float* ef1   = (const float*)d_in[4];
    const float* kqvW  = (const float*)d_in[5];
    const float* multb = (const float*)d_in[6];
    const float* addb  = (const float*)d_in[7];
    const float* efW   = (const float*)d_in[8];
    const float* combW = (const float*)d_in[9];
    const float* pre_g = (const float*)d_in[10];
    const float* pre_b = (const float*)d_in[11];
    const float* rz    = (const float*)d_in[12];
    const float* ffn_g = (const float*)d_in[13];
    const float* ffn_b = (const float*)d_in[14];
    const float* ff1W  = (const float*)d_in[15];
    const float* ff2W  = (const float*)d_in[16];
    const float* rz_ff = (const float*)d_in[17];
    float* out = (float*)d_out;

    const int V = in_sizes[0] / 256;
    const int E = in_sizes[1] / 2;

    // ---- workspace carve-out
    char* ws = (char*)d_ws;
    size_t off = 0;
    auto carve = [&](size_t bytes) { void* p = ws + off; off = ws_align(off + bytes); return p; };

    _Float16* xh      = (_Float16*)carve((size_t)V * 256 * 2);
    _Float16* kqvW16  = (_Float16*)carve((size_t)256 * 768 * 2);
    float*    kqv     = (float*)   carve((size_t)V * 768 * 4);
    _Float16* combW16 = (_Float16*)carve((size_t)128 * 256 * 2);
    _Float16* ff1W16  = (_Float16*)carve((size_t)256 * 512 * 2);
    _Float16* ff2W16  = (_Float16*)carve((size_t)512 * 256 * 2);
    float*    scores  = (float*)   carve((size_t)2 * E * 8 * 4);
    unsigned* smax    = (unsigned*)carve((size_t)V * 8 * 4);
    float*    den     = (float*)   carve((size_t)V * 8 * 4);
    float*    agg     = (float*)   carve((size_t)V * 128 * 4);
    _Float16* agg16   = (_Float16*)carve((size_t)V * 128 * 2);
    float*    out_tmp = (float*)   carve((size_t)V * 256 * 4);
    _Float16* y16     = (_Float16*)carve((size_t)V * 256 * 2);
    _Float16* hid16   = (_Float16*)carve((size_t)V * 512 * 2);
    (void)ws_size;

    const int TB = 256;
    const long long nAgg = (long long)V * 128;

    // ---- init softmax/agg buffers
    init_softmax_bufs<<<(unsigned)((nAgg + TB - 1) / TB), TB, 0, stream>>>(
        smax, den, agg, (long long)V * 8, nAgg);

    // ---- weight casts f32 -> f16
    cast_f32_to_f16<<<(256 * 768 + TB - 1) / TB, TB, 0, stream>>>(kqvW, kqvW16, 256LL * 768);
    cast_f32_to_f16<<<(128 * 256 + TB - 1) / TB, TB, 0, stream>>>(combW, combW16, 128LL * 256);
    cast_f32_to_f16<<<(256 * 512 + TB - 1) / TB, TB, 0, stream>>>(ff1W, ff1W16, 256LL * 512);
    cast_f32_to_f16<<<(512 * 256 + TB - 1) / TB, TB, 0, stream>>>(ff2W, ff2W16, 512LL * 256);

    // ---- LN1 -> x (f16)
    ln_f16_kernel<<<(V + 7) / 8, TB, 0, stream>>>(h_emb, pre_g, pre_b, xh, V);

    const int gm = (V + 63) / 64;

    // ---- kqv = x @ kqv_W  (V x 768, K=256)
    gemm_f16_wmma<0><<<dim3(gm, 768 / 64), TB, 0, stream>>>(
        xh, kqvW16, V, 768, 256, kqv, nullptr, nullptr, nullptr);

    // ---- edge attention (segment softmax over targets)
    edge_scores_kernel<<<E, TB, 0, stream>>>(adj0, adj1, ef0, ef1, kqv, multb,
                                             addb, efW, scores, smax, E);
    edge_exp_kernel<<<(unsigned)(((long long)2 * E * 8 + TB - 1) / TB), TB, 0, stream>>>(
        adj0, adj1, scores, smax, den, E);
    edge_agg_kernel<<<(unsigned)(((long long)2 * E * 128 + TB - 1) / TB), TB, 0, stream>>>(
        adj0, adj1, scores, den, kqv, agg, E);

    // ---- out_tmp = h + rz * (agg @ comb_W)   (K=128, N=256)
    cast_f32_to_f16<<<(unsigned)((nAgg + TB - 1) / TB), TB, 0, stream>>>(agg, agg16, nAgg);
    gemm_f16_wmma<1><<<dim3(gm, 256 / 64), TB, 0, stream>>>(
        agg16, combW16, V, 256, 128, out_tmp, nullptr, h_emb, rz);

    // ---- FFN: y = LN(out_tmp); hid = gelu(y @ ff1_W); out = out_tmp + rz_ff*(hid @ ff2_W)
    ln_f16_kernel<<<(V + 7) / 8, TB, 0, stream>>>(out_tmp, ffn_g, ffn_b, y16, V);
    gemm_f16_wmma<2><<<dim3(gm, 512 / 64), TB, 0, stream>>>(
        y16, ff1W16, V, 512, 256, nullptr, hid16, nullptr, nullptr);
    gemm_f16_wmma<1><<<dim3(gm, 256 / 64), TB, 0, stream>>>(
        hid16, ff2W16, V, 256, 512, out, nullptr, out_tmp, rz_ff);
}
